// LoRAGraphSelfAttention_26319559590753
// MI455X (gfx1250) — 16-bit and 8-bit matrix operand order repaired and checked by reading, compiled, not run on hardware
//
#include <hip/hip_runtime.h>
#include <hip/hip_bf16.h>
#include <math.h>

typedef __attribute__((ext_vector_type(16))) _Float16 v16h;
typedef __attribute__((ext_vector_type(8)))  float    v8f;
typedef __attribute__((ext_vector_type(8)))  int      v8i;

#define BDIM   8
#define NDIM   1024
#define INDIM  512
#define OUTDIM 512
#define HDIM   8
#define RDIM   8
#define HD     64
#define SCALING 2.0f
#define NEGV   (-1e9f)

static constexpr size_t QKV_ELEMS = (size_t)BDIM * HDIM * NDIM * HD;

static constexpr unsigned LDS_SCORES = 0;
static constexpr unsigned LDS_K8     = 65536;
static constexpr unsigned LDS_VCHUNK = 131072;
static constexpr unsigned LDS_TOTAL  = 135168;

__global__ void lga_weff(const float* __restrict__ Wq, const float* __restrict__ Wk,
                         const float* __restrict__ Wv, const float* __restrict__ Aq,
                         const float* __restrict__ Bq, const float* __restrict__ Ak,
                         const float* __restrict__ Bk, const float* __restrict__ Av,
                         const float* __restrict__ Bv,
                         _Float16* __restrict__ weff, unsigned* __restrict__ amax) {
  size_t gid = (size_t)blockIdx.x * blockDim.x + threadIdx.x;
  if (gid < 2) amax[gid] = 0u;
  if (gid >= (size_t)3 * INDIM * OUTDIM) return;
  int proj = (int)(gid / (INDIM * OUTDIM));
  int rem  = (int)(gid % (INDIM * OUTDIM));
  int i = rem / OUTDIM, j = rem % OUTDIM;
  const float* W  = proj == 0 ? Wq : (proj == 1 ? Wk : Wv);
  const float* A  = proj == 0 ? Aq : (proj == 1 ? Ak : Av);
  const float* Bm = proj == 0 ? Bq : (proj == 1 ? Bk : Bv);
  float acc = W[(size_t)i * OUTDIM + j];
#pragma unroll
  for (int r = 0; r < RDIM; ++r)
    acc += SCALING * A[(size_t)i * RDIM + r] * Bm[(size_t)r * OUTDIM + j];
  weff[gid] = (_Float16)acc;
}

__global__ void __launch_bounds__(256) lga_proj(const float* __restrict__ x,
                                                const _Float16* __restrict__ weff,
                                                _Float16* __restrict__ qkvh,
                                                unsigned* __restrict__ amax) {
  int wave = threadIdx.x >> 5, lane = threadIdx.x & 31;
  int tile = blockIdx.x * 8 + wave;
  int proj = tile / (512 * 32);
  int rem  = tile - proj * (512 * 32);
  int mt = rem >> 5, nt = rem & 31;
  int hi = lane >> 4, l15 = lane & 15;

  const _Float16* wp = weff + (size_t)proj * INDIM * OUTDIM;
  size_t xrow = (size_t)(mt * 16 + l15) * INDIM;
  int col = nt * 16 + l15;

  v8f acc = {};
  for (int kc = 0; kc < INDIM / 32; ++kc) {
    int k0 = kc * 32;
    v16h a, bf;
#pragma unroll
    for (int v = 0; v < 8; ++v) {
#pragma unroll
      for (int j = 0; j < 2; ++j) {
        int Ka = 16 * (v >> 2) + 8 * hi + 2 * (v & 3) + j;
        a[2 * v + j] = (_Float16)x[xrow + k0 + Ka];
        int Kb = 16 * (v >> 2) + 8 * hi + 2 * (v & 3) + j;
        bf[2 * v + j] = wp[(size_t)(k0 + Kb) * OUTDIM + col];
      }
    }
    acc = __builtin_amdgcn_wmma_f32_16x16x32_f16(false, a, false, bf,
                                                 (short)0, acc, false, false);
  }

  int h = col >> 6, d = col & 63;
  float am = 0.0f;
  _Float16* dst = qkvh + (size_t)proj * QKV_ELEMS;
#pragma unroll
  for (int v = 0; v < 8; ++v) {
    int m  = mt * 16 + v + 8 * hi;
    int bb = m >> 10, n = m & 1023;
    float val = acc[v];
    dst[(((size_t)bb * HDIM + h) * NDIM + n) * HD + d] = (_Float16)val;
    am = fmaxf(am, fabsf(val));
  }
  if (proj < 2) atomicMax(&amax[proj], __float_as_uint(am));
}

__global__ void lga_quant(const _Float16* __restrict__ qkvh,
                          const unsigned* __restrict__ amax,
                          signed char* __restrict__ q8) {
  size_t gid = (size_t)blockIdx.x * blockDim.x + threadIdx.x;
  int sel = (int)(gid >> 22);
  float sc = fmaxf(__uint_as_float(amax[sel]), 1e-8f) * (1.0f / 127.0f);
  float v = (float)qkvh[gid];
  float q = rintf(v / sc);
  q = fminf(fmaxf(q, -127.0f), 127.0f);
  q8[gid] = (signed char)q;
}

__global__ void __launch_bounds__(32) lga_attn(const signed char* __restrict__ q8,
                                               const _Float16* __restrict__ vh,
                                               const int* __restrict__ adj,
                                               const unsigned* __restrict__ amax,
                                               float* __restrict__ outp,
                                               float* __restrict__ pp) {
  extern __shared__ char smraw[];
  float*       sm  = (float*)(smraw + LDS_SCORES);
  signed char* kst = (signed char*)(smraw + LDS_K8);
  _Float16*    vst = (_Float16*)(smraw + LDS_VCHUNK);

  int lane = threadIdx.x;
  int bid  = blockIdx.x;
  int mt = bid & 63;
  int h  = (bid >> 6) & 7;
  int b  = bid >> 9;
  int hi = lane >> 4, l15 = lane & 15;
  int row0 = mt * 16;
  const signed char* k8 = q8 + QKV_ELEMS;
  size_t bh = ((size_t)b * HDIM + h) * NDIM;

  {
    const signed char* kg = k8 + bh * HD;
    unsigned lbase = (unsigned)(size_t)kst;
    for (int it = 0; it < 128; ++it) {
      unsigned off = (unsigned)it * 512u + (unsigned)lane * 16u;
      unsigned long long ga = (unsigned long long)(size_t)(kg + off);
      unsigned la = lbase + off;
      asm volatile("global_load_async_to_lds_b128 %0, %1, off"
                   :: "v"(la), "v"(ga) : "memory");
    }
    asm volatile("s_wait_asynccnt 0" ::: "memory");
  }

  float sq = fmaxf(__uint_as_float(amax[0]), 1e-8f) * (1.0f / 127.0f);
  float sk = fmaxf(__uint_as_float(amax[1]), 1e-8f) * (1.0f / 127.0f);
  float ss = sq * sk;

  v8i afrag;
  {
    size_t base = (bh + row0 + l15) * HD;
#pragma unroll
    for (int v = 0; v < 8; ++v) {
      int Ka = 32 * (v >> 2) + 16 * ((v >> 1) & 1) + 8 * hi + 4 * (v & 1);
      afrag[v] = *(const int*)(q8 + base + Ka);
    }
  }

  v8i czero = {};
  for (int ct = 0; ct < 64; ++ct) {
    unsigned kb = (unsigned)(ct * 16 + l15) * HD;
    v8i bfrag;
#pragma unroll
    for (int v = 0; v < 8; ++v) {
      int Kb = 32 * (v >> 2) + 16 * ((v >> 1) & 1) + 8 * hi + 4 * (v & 1);
      bfrag[v] = *(const int*)(kst + kb + Kb);
    }
    if (ct + 1 < 64)
      __builtin_prefetch(adj + ((size_t)b * NDIM + row0) * NDIM + (ct + 1) * 16, 0, 0);

    v8i iacc = __builtin_amdgcn_wmma_i32_16x16x64_iu8(true, afrag, true, bfrag,
                                                      czero, false, false);
#pragma unroll
    for (int v = 0; v < 8; ++v) {
      int M = v + 8 * hi;
      int colc = ct * 16 + l15;
      float sval = (float)iacc[v] * ss;
      int a = adj[((size_t)b * NDIM + row0 + M) * NDIM + colc];
      sm[M * NDIM + colc] = (a == 0) ? NEGV : sval;
    }
  }
  __syncthreads();

  if (lane < 16) {
    float* row = sm + lane * NDIM;
    float mx = -3.4e38f;
    for (int j = 0; j < NDIM; ++j) mx = fmaxf(mx, row[j]);
    float s = 0.0f;
    for (int j = 0; j < NDIM; ++j) { float e = __expf(row[j] - mx); row[j] = e; s += e; }
    float inv = 1.0f / s;
    size_t pbase = (bh + row0 + lane) * NDIM;
    for (int j = 0; j < NDIM; ++j) {
      float pq = rintf(row[j] * inv * 255.0f) * (1.0f / 255.0f);
      row[j] = pq;
      pp[pbase + j] = pq;
    }
  }
  __syncthreads();

  v8f zero = {};
  v8f acc0 = zero, acc1 = zero, acc2 = zero, acc3 = zero;
  unsigned vbase = (unsigned)(size_t)vst;
  for (int kc = 0; kc < NDIM / 32; ++kc) {
    int k0 = kc * 32;

    {
      const char* vg = (const char*)(vh + (bh + k0) * HD);
#pragma unroll
      for (int it = 0; it < 8; ++it) {
        unsigned off = (unsigned)it * 512u + (unsigned)lane * 16u;
        unsigned long long ga = (unsigned long long)(size_t)(vg + off);
        asm volatile("global_load_async_to_lds_b128 %0, %1, off"
                     :: "v"(vbase + off), "v"(ga) : "memory");
      }
      asm volatile("s_wait_asynccnt 0" ::: "memory");
    }

    v16h a;
#pragma unroll
    for (int v = 0; v < 8; ++v) {
#pragma unroll
      for (int j = 0; j < 2; ++j) {
        int Ka = 16 * (v >> 2) + 8 * hi + 2 * (v & 3) + j;
        a[2 * v + j] = (_Float16)sm[l15 * NDIM + k0 + Ka];
      }
    }
#pragma unroll
    for (int dt = 0; dt < 4; ++dt) {
      v16h bf;
#pragma unroll
      for (int v = 0; v < 8; ++v) {
#pragma unroll
        for (int j = 0; j < 2; ++j) {
          int Kb = 16 * (v >> 2) + 8 * hi + 2 * (v & 3) + j;
          bf[2 * v + j] = vst[Kb * HD + dt * 16 + l15];
        }
      }
      v8f& accr = dt == 0 ? acc0 : dt == 1 ? acc1 : dt == 2 ? acc2 : acc3;
      accr = __builtin_amdgcn_wmma_f32_16x16x32_f16(false, a, false, bf,
                                                    (short)0, accr, false, false);
    }
  }
#pragma unroll
  for (int dt = 0; dt < 4; ++dt) {
    v8f accr = dt == 0 ? acc0 : dt == 1 ? acc1 : dt == 2 ? acc2 : acc3;
#pragma unroll
    for (int v = 0; v < 8; ++v) {
      int rw = row0 + v + 8 * hi;
      int cl = h * HD + dt * 16 + l15;
      outp[((size_t)b * NDIM + rw) * OUTDIM + cl] = accr[v];
    }
  }
}

extern "C" void kernel_launch(void* const* d_in, const int* in_sizes, int n_in,
                              void* d_out, int out_size, void* d_ws, size_t ws_size,
                              hipStream_t stream) {
  (void)in_sizes; (void)n_in; (void)out_size; (void)ws_size;
  const float* x  = (const float*)d_in[0];
  const int*  adj = (const int*)d_in[1];
  const float* Wq = (const float*)d_in[2];
  const float* Wk = (const float*)d_in[3];
  const float* Wv = (const float*)d_in[4];
  const float* Aq = (const float*)d_in[5];
  const float* Bq = (const float*)d_in[6];
  const float* Ak = (const float*)d_in[7];
  const float* Bk = (const float*)d_in[8];
  const float* Av = (const float*)d_in[9];
  const float* Bv = (const float*)d_in[10];

  float* outp = (float*)d_out;
  float* pp   = outp + (size_t)BDIM * NDIM * OUTDIM;

  char* wsb = (char*)d_ws;
  _Float16* weff = (_Float16*)wsb;
  _Float16* qkvh = (_Float16*)(wsb + (size_t)3 * INDIM * OUTDIM * sizeof(_Float16));
  signed char* q8 = (signed char*)((char*)qkvh + 3 * QKV_ELEMS * sizeof(_Float16));
  unsigned* amax = (unsigned*)(q8 + 2 * QKV_ELEMS);

  lga_weff<<<3072, 256, 0, stream>>>(Wq, Wk, Wv, Aq, Bq, Ak, Bk, Av, Bv, weff, amax);
  lga_proj<<<6144, 256, 0, stream>>>(x, weff, qkvh, amax);
  lga_quant<<<(unsigned)((2 * QKV_ELEMS) / 256), 256, 0, stream>>>(qkvh, amax, q8);
  lga_attn<<<BDIM * HDIM * (NDIM / 16), 32, LDS_TOTAL, stream>>>(
      q8, qkvh + 2 * QKV_ELEMS, adj, amax, outp, pp);
}
